// ImprovedGATV2_53463752900681
// MI455X (gfx1250) — matrix operand order repaired and checked by reading, not compiled, not run on hardware
//
#include <hip/hip_runtime.h>
#include <hip/hip_bf16.h>
#include <stdint.h>

// ---------------------------------------------------------------------------
// GATv2 forward for MI455X (gfx1250, wave32).
//  - GEMMs: v_wmma_f32_16x16x32_bf16, weights staged to LDS via the Tensor
//    Data Mover (tensor_load_to_lds + s_wait_tensorcnt), A streamed from HBM.
//  - Edge softmax/aggregation: f32 L2 atomics (scatter-bound phase).
// ---------------------------------------------------------------------------

typedef __bf16        v16bf __attribute__((ext_vector_type(16)));
typedef float         v8f   __attribute__((ext_vector_type(8)));
typedef unsigned int  v4u   __attribute__((ext_vector_type(4)));
typedef int           v8i   __attribute__((ext_vector_type(8)));
typedef int           v4i   __attribute__((ext_vector_type(4)));

union FragU { v16bf v; uint4 q[2]; };

#define GK 256            // inner dimension is 256 for every GEMM in this net
#define NEG_SLOPE 0.2f
#define LDS_ROW 264       // 256 bf16 + 8 bf16 pad (16B) -> bank-spread rows

// --- helpers ----------------------------------------------------------------

__device__ __forceinline__ unsigned short f2bf(float f) {
  unsigned int u = __float_as_uint(f);
  u += 0x7fffu + ((u >> 16) & 1u);          // round-to-nearest-even
  return (unsigned short)(u >> 16);
}

// order-preserving float <-> int key for atomicMax-based segment max
__device__ __forceinline__ int fkey(float f) {
  int i = __float_as_int(f);
  return i >= 0 ? i : (i ^ 0x7fffffff);
}
__device__ __forceinline__ float funkey(int i) {
  return __int_as_float(i >= 0 ? i : (i ^ 0x7fffffff));
}

// --- conversion kernels -----------------------------------------------------

__global__ void k_cvt_bf16(const float* __restrict__ src,
                           unsigned short* __restrict__ dst, int n) {
  int i = blockIdx.x * blockDim.x + threadIdx.x;
  if (i < n) dst[i] = f2bf(src[i]);
}

// WT[n*K + k] = bf16(W[k*Nc + n])   (transpose so B fragments are K-contiguous)
__global__ void k_cvt_w_t(const float* __restrict__ W,
                          unsigned short* __restrict__ WT, int K, int Nc) {
  int i = blockIdx.x * blockDim.x + threadIdx.x;
  if (i >= K * Nc) return;
  int k = i / Nc;
  int n = i - k * Nc;
  WT[(size_t)n * K + k] = f2bf(W[i]);
}

// --- WMMA GEMM: C[M,Nc] = A[M,GK] (bf16) * B[GK,Nc] (bf16, given as BT[Nc,GK])
//
// Grid: each block owns one 64-column strip of 8 consecutive 16-row tiles.
// Block prologue: wave 0 issues ONE Tensor-DMA moving the strip
// BT[cb*64 .. cb*64+64) x GK (contiguous 32KB) into LDS, with TDM padding
// (16B per 512B row) so ds_load_b128 fragment reads are bank-spread.
//
// Fragment layouts used here (contiguous k, same order for A and B):
//   A 16x32 bf16 : lane(l&15)=row, halves 0..7  = K[(l>>4)*16 .. +8),
//                                halves 8..15 = K[(l>>4)*16+8 .. +8)
//   B 32x16 bf16 : lane(l&15)=col, halves 0..15 = K[(l>>4)*16 .. +16)
// ---------------------------------------------------------------------------

__global__ __launch_bounds__(256)
void k_gemm_bf16(const unsigned short* __restrict__ A,
                 const unsigned short* __restrict__ BT,
                 float* __restrict__ C, int M, int Nc) {
  __shared__ unsigned short ldsB[64 * LDS_ROW];   // 33792 B

  const int colBlocks = Nc >> 6;                  // 64-column strips
  const int bid    = (int)blockIdx.x;
  const int cb     = bid % colBlocks;
  const int rbBase = (bid / colBlocks) * 8;
  const int MT     = M >> 4;

  if (threadIdx.x < 32) {
    // ---- Tensor DMA descriptor (D#), ISA 8.3/8.4 ----
    const unsigned elems = 64u * GK;              // 16384 x 2B = 32KB strip
    unsigned ldsAddr = (unsigned)(size_t)&ldsB[0];            // low 32b = LDS offset
    unsigned long long ga =
        (unsigned long long)(size_t)(BT + (size_t)cb * 64 * GK);

    v4u g0;
    g0[0] = 1u;                                   // count=1, user descriptor
    g0[1] = ldsAddr;                              // lds_addr
    g0[2] = (unsigned)(ga & 0xffffffffu);         // global_addr[31:0]
    g0[3] = (unsigned)((ga >> 32) & 0x01ffffffu)  // global_addr[56:32]
          | (2u << 30);                           // type = 2 ("image")

    // group1: data_size=2B, pad_enable, pad every 128 DWORDs (512B) by 4 DWORDs
    unsigned w0 = (1u << 16) | (1u << 20) | (6u << 22) | (3u << 25);
    unsigned w1 = (elems & 0xffffu) << 16;        // tensor_dim0[15:0]
    unsigned w2 = (elems >> 16) | (1u << 16);     // tensor_dim0[31:16], tensor_dim1=1
    unsigned w3 = (elems & 0xffffu) << 16;        // tile_dim0
    unsigned w4 = 1u;                             // tile_dim1=1, tile_dim2=0
    unsigned w5 = elems;                          // tensor_dim0_stride[31:0]
    v8i g1;
    g1[0] = (int)w0; g1[1] = (int)w1; g1[2] = (int)w2; g1[3] = (int)w3;
    g1[4] = (int)w4; g1[5] = (int)w5; g1[6] = 0;  g1[7] = 0;

    v4i gz4 = {0, 0, 0, 0};                       // <=2D tensor: groups 2/3 unused
    v8i gz8 = {0, 0, 0, 0, 0, 0, 0, 0};
    __builtin_amdgcn_tensor_load_to_lds(g0, g1, gz4, gz4, gz8, 0);
    __builtin_amdgcn_s_wait_tensorcnt(0);
  }
  __syncthreads();                                // LDS strip visible to all waves

  const int wv   = (int)(threadIdx.x >> 5);
  const int lane = (int)(threadIdx.x & 31);
  const int rb   = rbBase + wv;
  if (rb < MT) {
    const int l15 = lane & 15;
    const int hi  = lane >> 4;

    const unsigned short* aRow = A + (size_t)(rb * 16 + l15) * GK + hi * 16;
    const unsigned short* lb0  = &ldsB[(l15)      * LDS_ROW + hi * 16];
    const unsigned short* lb1  = &ldsB[(16 + l15) * LDS_ROW + hi * 16];
    const unsigned short* lb2  = &ldsB[(32 + l15) * LDS_ROW + hi * 16];
    const unsigned short* lb3  = &ldsB[(48 + l15) * LDS_ROW + hi * 16];

    v8f acc0 = {}, acc1 = {}, acc2 = {}, acc3 = {};

#pragma unroll
    for (int k0 = 0; k0 < GK; k0 += 32) {
      FragU a, b0, b1, b2, b3;
      a.q[0]  = *(const uint4*)(aRow + k0);
      a.q[1]  = *(const uint4*)(aRow + k0 + 8);
      b0.q[0] = *(const uint4*)(lb0 + k0);
      b0.q[1] = *(const uint4*)(lb0 + k0 + 8);
      b1.q[0] = *(const uint4*)(lb1 + k0);
      b1.q[1] = *(const uint4*)(lb1 + k0 + 8);
      b2.q[0] = *(const uint4*)(lb2 + k0);
      b2.q[1] = *(const uint4*)(lb2 + k0 + 8);
      b3.q[0] = *(const uint4*)(lb3 + k0);
      b3.q[1] = *(const uint4*)(lb3 + k0 + 8);

      acc0 = __builtin_amdgcn_wmma_f32_16x16x32_bf16(false, a.v, false, b0.v,
                                                     (short)0, acc0, false, false);
      acc1 = __builtin_amdgcn_wmma_f32_16x16x32_bf16(false, a.v, false, b1.v,
                                                     (short)0, acc1, false, false);
      acc2 = __builtin_amdgcn_wmma_f32_16x16x32_bf16(false, a.v, false, b2.v,
                                                     (short)0, acc2, false, false);
      acc3 = __builtin_amdgcn_wmma_f32_16x16x32_bf16(false, a.v, false, b3.v,
                                                     (short)0, acc3, false, false);
    }

    // C/D layout: lane(l&15)=col, VGPR i = row (l>>4)*8 + i
    const int col0 = cb * 64 + l15;
    const int row0 = rb * 16 + hi * 8;
#pragma unroll
    for (int i = 0; i < 8; ++i) {
      float* crow = C + (size_t)(row0 + i) * Nc + col0;
      crow[0]  = acc0[i];
      crow[16] = acc1[i];
      crow[32] = acc2[i];
      crow[48] = acc3[i];
    }
  }
}

// --- per-layer state init ---------------------------------------------------

__global__ void k_init(int* __restrict__ mx, float* __restrict__ denom,
                       float* __restrict__ agg, int nH, int nHC) {
  int i = blockIdx.x * blockDim.x + threadIdx.x;
  if (i < nHC) agg[i] = 0.f;
  if (i < nH)  { mx[i] = (int)0x80000000; denom[i] = 0.f; }
}

// --- edge kernels (one thread per (edge, head)) -----------------------------

__global__ void k_edge_logits(const float* __restrict__ xl, const float* __restrict__ xr,
                              const float* __restrict__ att, const int* __restrict__ ei,
                              int E, int ET, int H, int C,
                              float* __restrict__ logits, int* __restrict__ mx) {
  int t = blockIdx.x * blockDim.x + threadIdx.x;
  if (t >= ET * H) return;
  int e = t / H, h = t - e * H;
  int s, d;
  if (e < E) { s = ei[e]; d = ei[E + e]; } else { s = e - E; d = s; }  // self loops
  const int HC = H * C;
  const float* pl = xl + (size_t)s * HC + h * C;
  const float* pr = xr + (size_t)d * HC + h * C;
  const float* pa = att + h * C;
  __builtin_prefetch(pl, 0, 0);
  __builtin_prefetch(pr, 0, 0);
  float acc = 0.f;
  for (int c = 0; c < C; c += 4) {
    float4 a = *(const float4*)(pl + c);
    float4 b = *(const float4*)(pr + c);
    float4 w = *(const float4*)(pa + c);
    float v;
    v = a.x + b.x; v = v > 0.f ? v : NEG_SLOPE * v; acc += w.x * v;
    v = a.y + b.y; v = v > 0.f ? v : NEG_SLOPE * v; acc += w.y * v;
    v = a.z + b.z; v = v > 0.f ? v : NEG_SLOPE * v; acc += w.z * v;
    v = a.w + b.w; v = v > 0.f ? v : NEG_SLOPE * v; acc += w.w * v;
  }
  logits[t] = acc;
  atomicMax(mx + (size_t)d * H + h, fkey(acc));
}

__global__ void k_edge_expsum(float* __restrict__ logits, const int* __restrict__ mx,
                              const int* __restrict__ ei, int E, int ET, int H,
                              float* __restrict__ denom) {
  int t = blockIdx.x * blockDim.x + threadIdx.x;
  if (t >= ET * H) return;
  int e = t / H, h = t - e * H;
  int d = (e < E) ? ei[E + e] : (e - E);
  float m  = funkey(mx[(size_t)d * H + h]);
  float ex = __expf(logits[t] - m);
  logits[t] = ex;                       // reuse buffer for exp values
  atomicAdd(denom + (size_t)d * H + h, ex);
}

__global__ void k_edge_agg(const float* __restrict__ ex, const float* __restrict__ denom,
                           const float* __restrict__ xl, const int* __restrict__ ei,
                           int E, int ET, int H, int C, float* __restrict__ agg) {
  int t = blockIdx.x * blockDim.x + threadIdx.x;
  if (t >= ET * H) return;
  int e = t / H, h = t - e * H;
  int s, d;
  if (e < E) { s = ei[e]; d = ei[E + e]; } else { s = e - E; d = s; }
  float alpha = ex[t] / denom[(size_t)d * H + h];
  const int HC = H * C;
  const float* pl = xl  + (size_t)s * HC + h * C;
  float*       pa = agg + (size_t)d * HC + h * C;
  __builtin_prefetch(pl, 0, 0);
  for (int c = 0; c < C; c += 4) {
    float4 v = *(const float4*)(pl + c);
    atomicAdd(pa + c + 0, alpha * v.x);
    atomicAdd(pa + c + 1, alpha * v.y);
    atomicAdd(pa + c + 2, alpha * v.z);
    atomicAdd(pa + c + 3, alpha * v.w);
  }
}

// --- bias + activation ------------------------------------------------------

__global__ void k_finalize(const float* __restrict__ agg, const float* __restrict__ bias,
                           float* __restrict__ dst, int n, int HC, int doElu) {
  int i = blockIdx.x * blockDim.x + threadIdx.x;
  if (i >= n) return;
  float v = agg[i] + bias[i % HC];
  if (doElu) v = v > 0.f ? v : (__expf(v) - 1.f);
  dst[i] = v;
}

// ---------------------------------------------------------------------------

extern "C" void kernel_launch(void* const* d_in, const int* in_sizes, int n_in,
                              void* d_out, int out_size, void* d_ws, size_t ws_size,
                              hipStream_t stream) {
  (void)n_in; (void)out_size; (void)ws_size;

  const float* x       = (const float*)d_in[0];
  const int*   ei      = (const int*)d_in[1];     // JAX default: int32 (x64 disabled)
  const float* Wl[3]   = {(const float*)d_in[2],  (const float*)d_in[6],  (const float*)d_in[10]};
  const float* Wr[3]   = {(const float*)d_in[3],  (const float*)d_in[7],  (const float*)d_in[11]};
  const float* att[3]  = {(const float*)d_in[4],  (const float*)d_in[8],  (const float*)d_in[12]};
  const float* bias[3] = {(const float*)d_in[5],  (const float*)d_in[9],  (const float*)d_in[13]};

  const int N  = in_sizes[0] / 256;   // 50000
  const int E  = in_sizes[1] / 2;     // 800000
  const int ET = E + N;               // + self loops

  // --- carve workspace (256B aligned slabs) ---
  char* p = (char*)d_ws;
  auto take = [&](size_t bytes) -> void* {
    void* r = (void*)p;
    p += (bytes + 255) & ~(size_t)255;
    return r;
  };
  unsigned short* actb = (unsigned short*)take((size_t)N * 256 * 2);
  unsigned short* wlT  = (unsigned short*)take((size_t)256 * 256 * 2);
  unsigned short* wrT  = (unsigned short*)take((size_t)256 * 256 * 2);
  float* xl     = (float*)take((size_t)N * 256 * 4);
  float* xr     = (float*)take((size_t)N * 256 * 4);
  float* agg    = (float*)take((size_t)N * 256 * 4);
  float* logits = (float*)take((size_t)ET * 8 * 4);
  int*   mx     = (int*)  take((size_t)N * 8 * 4);
  float* denom  = (float*)take((size_t)N * 8 * 4);

  struct Cfg { int Nc, H, C, elu; };
  const Cfg L[3] = { {256, 8, 32, 1}, {256, 8, 32, 1}, {64, 1, 64, 0} };

  auto grid = [](int n) { return dim3((unsigned)((n + 255) / 256)); };

  const float* actsrc = x;
  for (int l = 0; l < 3; ++l) {
    const int Nc = L[l].Nc, H = L[l].H, C = L[l].C;

    // bf16 conversion (activations + transposed weights)
    k_cvt_bf16<<<grid(N * GK), 256, 0, stream>>>(actsrc, actb, N * GK);
    k_cvt_w_t <<<grid(GK * Nc), 256, 0, stream>>>(Wl[l], wlT, GK, Nc);
    k_cvt_w_t <<<grid(GK * Nc), 256, 0, stream>>>(Wr[l], wrT, GK, Nc);

    // WMMA GEMMs: xl = act @ Wl, xr = act @ Wr
    // one block = one 64-col strip x 8 row-tiles, weights staged via TDM
    const int MT     = N / 16;
    const int blocks = ((MT + 7) / 8) * (Nc / 64);
    k_gemm_bf16<<<blocks, 256, 0, stream>>>(actb, wlT, xl, N, Nc);
    k_gemm_bf16<<<blocks, 256, 0, stream>>>(actb, wrT, xr, N, Nc);

    // softmax state init
    k_init<<<grid(N * Nc), 256, 0, stream>>>(mx, denom, agg, N * H, N * Nc);

    // edge phase: logits + segment max -> exp + segment sum -> weighted scatter
    const int eth = ET * H;
    k_edge_logits<<<grid(eth), 256, 0, stream>>>(xl, xr, att[l], ei, E, ET, H, C, logits, mx);
    k_edge_expsum<<<grid(eth), 256, 0, stream>>>(logits, mx, ei, E, ET, H, denom);
    k_edge_agg   <<<grid(eth), 256, 0, stream>>>(logits, denom, xl, ei, E, ET, H, C, agg);

    // bias + ELU (layers 0,1 write in-place and feed the next layer)
    float* dst = (l == 2) ? (float*)d_out : agg;
    k_finalize<<<grid(N * Nc), 256, 0, stream>>>(agg, bias[l], dst, N * Nc, Nc, L[l].elu);

    actsrc = agg;
  }
}
